// H2GCN_47991964565814
// MI455X (gfx1250) — compile-verified, matrix operand order checked by reading, not run on hardware
//
#include <hip/hip_runtime.h>
#include <stdint.h>

#define NN    8192
#define NE    131072
#define FEAT  512
#define HID   256
#define OUTD  1792
#define NG    64

typedef __attribute__((ext_vector_type(16))) __bf16 v16bf;
typedef __attribute__((ext_vector_type(8)))  float  v8f;
typedef __attribute__((ext_vector_type(8)))  int    v8i;

union BF16Frag { v16bf v; unsigned u[8]; };
union I8Frag   { v8i   v; unsigned u[8]; };

__device__ __forceinline__ unsigned short f32_to_bf16(float f) {
    union { float f; unsigned u; } c; c.f = f;
    unsigned r = c.u + 0x7FFFu + ((c.u >> 16) & 1u);
    return (unsigned short)(r >> 16);
}

__device__ __forceinline__ void async_cp16(unsigned lds, const void* g) {
    asm volatile("global_load_async_to_lds_b128 %0, %1, off"
                 :: "v"(lds), "v"((unsigned long long)(size_t)g) : "memory");
}
__device__ __forceinline__ void async_cp16_o16(unsigned lds, const void* g) {
    asm volatile("global_load_async_to_lds_b128 %0, %1, off offset:16"
                 :: "v"(lds), "v"((unsigned long long)(size_t)g) : "memory");
}
__device__ __forceinline__ void wait_async3() {
    asm volatile("s_wait_asynccnt 0x3" ::: "memory");
}
__device__ __forceinline__ void wait_async0() {
    asm volatile("s_wait_asynccnt 0x0" ::: "memory");
}

__global__ void k_zero_u32(unsigned* __restrict__ p, long long n) {
    long long i = blockIdx.x * (long long)blockDim.x + threadIdx.x;
    long long st = (long long)gridDim.x * blockDim.x;
    for (; i < n; i += st) p[i] = 0u;
}

__global__ void k_scatter(const int* __restrict__ e,
                          unsigned char* __restrict__ Ab,
                          unsigned char* __restrict__ AbT,
                          unsigned* __restrict__ selfdup) {
    int i = blockIdx.x * blockDim.x + threadIdx.x;
    if (i >= NE) return;
    unsigned r = (unsigned)e[i], c = (unsigned)e[NE + i];
    if (r < NN && c < NN) {
        Ab [(size_t)r * NN + c] = 1;
        AbT[(size_t)c * NN + r] = 1;
        if (r == c) atomicAdd(&selfdup[r], 1u);
    }
}

__global__ __launch_bounds__(256)
void k_adj2(const unsigned char* __restrict__ Ab,
            const unsigned char* __restrict__ AbT,
            const unsigned* __restrict__ selfdup,
            unsigned short* __restrict__ A1,
            unsigned short* __restrict__ A2) {
    __shared__ __align__(16) unsigned As[2][128 * 20];
    __shared__ __align__(16) unsigned Bs[2][64 * 20];

    const int tid = threadIdx.x;
    const int rowbase = blockIdx.x * 128;
    const int colbase = blockIdx.y * 64;
    const int lane = tid & 31, wave = tid >> 5;
    const int wm = wave & 3, wn = wave >> 2;
    const int g = lane >> 4, lr = lane & 15;

    const int ar = tid >> 1, ap = tid & 1;
    const int bcc = tid >> 2, bch = tid & 3;

    const unsigned ldsA = (unsigned)(size_t)&As[0][0] + (unsigned)(ar * 80 + ap * 32);
    const unsigned ldsB = (unsigned)(size_t)&Bs[0][0] + (unsigned)(bcc * 80 + bch * 16);
    const unsigned strA = 128 * 20 * 4, strB = 64 * 20 * 4;
    const unsigned char* gA = Ab  + (size_t)(rowbase + ar) * NN + ap * 32;
    const unsigned char* gB = AbT + (size_t)(colbase + bcc) * NN + bch * 16;

    I8Frag acc[2][2];
#pragma unroll
    for (int a = 0; a < 2; ++a)
#pragma unroll
        for (int b = 0; b < 2; ++b)
#pragma unroll
            for (int k = 0; k < 8; ++k) acc[a][b].u[k] = 0u;

    async_cp16(ldsA, gA);
    async_cp16_o16(ldsA, gA);
    async_cp16(ldsB, gB);

    int buf = 0;
    for (int k0 = 0; k0 < NN; k0 += 64) {
        int nk = k0 + 64;
        if (nk < NN) {
            unsigned la = ldsA + (unsigned)(buf ^ 1) * strA;
            unsigned lb = ldsB + (unsigned)(buf ^ 1) * strB;
            async_cp16(la, gA + nk);
            async_cp16_o16(la, gA + nk);
            async_cp16(lb, gB + nk);
            wait_async3();
        } else {
            wait_async0();
        }
        __syncthreads();

        const unsigned* Al = &As[buf][0];
        const unsigned* Bl = &Bs[buf][0];
        I8Frag af[2], bfr[2];
#pragma unroll
        for (int mt = 0; mt < 2; ++mt) {
            int row = wm * 32 + mt * 16 + lr;
#pragma unroll
            for (int v = 0; v < 8; ++v) {
                int k = ((v >> 2) << 5) + (((v >> 1) & 1) << 4) + ((v & 1) << 2) + (g << 3);
                af[mt].u[v] = Al[row * 20 + (k >> 2)];
            }
        }
#pragma unroll
        for (int nt = 0; nt < 2; ++nt) {
            int col = wn * 32 + nt * 16 + lr;
#pragma unroll
            for (int v = 0; v < 8; ++v) {
                int k = ((v >> 2) << 5) + (((v >> 1) & 1) << 4) + ((v & 1) << 2) + (g << 3);
                bfr[nt].u[v] = Bl[col * 20 + (k >> 2)];
            }
        }
#pragma unroll
        for (int mt = 0; mt < 2; ++mt)
#pragma unroll
            for (int nt = 0; nt < 2; ++nt)
                acc[mt][nt].v = __builtin_amdgcn_wmma_i32_16x16x64_iu8(
                    false, af[mt].v, false, bfr[nt].v, acc[mt][nt].v, false, false);
        __syncthreads();
        buf ^= 1;
    }

#pragma unroll
    for (int mt = 0; mt < 2; ++mt)
#pragma unroll
        for (int nt = 0; nt < 2; ++nt)
#pragma unroll
            for (int c = 0; c < 8; ++c) {
                int row = rowbase + wm * 32 + mt * 16 + c + g * 8;
                int col = colbase + wn * 32 + nt * 16 + lr;
                size_t idx = (size_t)row * NN + col;
                unsigned ab = Ab[idx];
                int aa = acc[mt][nt].v[c];
                const unsigned short one = 0x3F80u;
                unsigned short o1 = (row != col) ? (ab ? one : (unsigned short)0)
                                                 : ((selfdup[row] >= 2u) ? one : (unsigned short)0);
                unsigned short o2 = (aa > 0 && ab == 0u && row != col) ? one : (unsigned short)0;
                A1[idx] = o1;
                A2[idx] = o2;
            }
}

__device__ __forceinline__ int nz2(unsigned x) {
    return ((x & 0xFFFFu) ? 1 : 0) + ((x >> 16) ? 1 : 0);
}

__global__ __launch_bounds__(256)
void k_degree(const unsigned short* __restrict__ A1,
              const unsigned short* __restrict__ A2,
              float* __restrict__ dinv1, float* __restrict__ dinv2) {
    __shared__ float s1[256], s2[256];
    int row = blockIdx.x, tid = threadIdx.x;
    const uint4* p1 = (const uint4*)(A1 + (size_t)row * NN);
    const uint4* p2 = (const uint4*)(A2 + (size_t)row * NN);
    int c1 = 0, c2 = 0;
    for (int i = tid; i < NN / 8; i += 256) {
        uint4 v = p1[i]; c1 += nz2(v.x) + nz2(v.y) + nz2(v.z) + nz2(v.w);
        uint4 w = p2[i]; c2 += nz2(w.x) + nz2(w.y) + nz2(w.z) + nz2(w.w);
    }
    s1[tid] = (float)c1; s2[tid] = (float)c2;
    __syncthreads();
    for (int s = 128; s > 0; s >>= 1) {
        if (tid < s) { s1[tid] += s1[tid + s]; s2[tid] += s2[tid + s]; }
        __syncthreads();
    }
    if (tid == 0) {
        float d1 = s1[0], d2 = s2[0];
        dinv1[row] = d1 > 0.f ? 1.0f / sqrtf(d1) : 0.0f;
        dinv2[row] = d2 > 0.f ? 1.0f / sqrtf(d2) : 0.0f;
    }
}

__global__ void k_cast_bf16(const float* __restrict__ s,
                            unsigned short* __restrict__ d, long long n) {
    long long i = blockIdx.x * (long long)blockDim.x + threadIdx.x;
    if (i < n) d[i] = f32_to_bf16(s[i]);
}

__global__ void k_castT_w(const float* __restrict__ w, unsigned short* __restrict__ wT) {
    int idx = blockIdx.x * blockDim.x + threadIdx.x;
    if (idx >= HID * FEAT) return;
    int c = idx / FEAT, k = idx % FEAT;
    wT[idx] = f32_to_bf16(w[(size_t)k * HID + c]);
}

__global__ void k_scale_castT(const float* __restrict__ h,
                              const float* __restrict__ dinv,
                              unsigned short* __restrict__ oT, int width, int coff) {
    long long idx = blockIdx.x * (long long)blockDim.x + threadIdx.x;
    long long total = (long long)NN * width;
    if (idx >= total) return;
    int j = (int)(idx / NN), i = (int)(idx % NN);
    oT[idx] = f32_to_bf16(dinv[i] * h[(size_t)i * OUTD + coff + j]);
}

__global__ __launch_bounds__(256)
void k_gemm_bf16(const unsigned short* __restrict__ A,
                 const unsigned short* __restrict__ BT,
                 float* __restrict__ out,
                 int K, int W, int ldc, int coff,
                 const float* __restrict__ scale, int relu) {
    __shared__ __align__(16) unsigned As[2][128 * 20];
    __shared__ __align__(16) unsigned Bs[2][64 * 20];

    const int tid = threadIdx.x;
    const int rowbase = blockIdx.x * 128;
    const int colbase = blockIdx.y * 64;
    const int lane = tid & 31, wave = tid >> 5;
    const int wm = wave & 3, wn = wave >> 2;
    const int g = lane >> 4, lr = lane & 15;

    const int ar = tid >> 1, ap = tid & 1;
    const int bcc = tid >> 2, bch = tid & 3;

    const unsigned ldsA = (unsigned)(size_t)&As[0][0] + (unsigned)(ar * 80 + ap * 32);
    const unsigned ldsB = (unsigned)(size_t)&Bs[0][0] + (unsigned)(bcc * 80 + bch * 16);
    const unsigned strA = 128 * 20 * 4, strB = 64 * 20 * 4;
    const unsigned char* gA = (const unsigned char*)A  + ((size_t)(rowbase + ar) * K) * 2 + ap * 32;
    const unsigned char* gB = (const unsigned char*)BT + ((size_t)(colbase + bcc) * K) * 2 + bch * 16;

    v8f acc[2][2];
#pragma unroll
    for (int a = 0; a < 2; ++a)
#pragma unroll
        for (int b = 0; b < 2; ++b)
#pragma unroll
            for (int c = 0; c < 8; ++c) acc[a][b][c] = 0.0f;

    async_cp16(ldsA, gA);
    async_cp16_o16(ldsA, gA);
    async_cp16(ldsB, gB);

    int buf = 0;
    for (int k0 = 0; k0 < K; k0 += 32) {
        int nk = k0 + 32;
        if (nk < K) {
            unsigned la = ldsA + (unsigned)(buf ^ 1) * strA;
            unsigned lb = ldsB + (unsigned)(buf ^ 1) * strB;
            async_cp16(la, gA + (size_t)nk * 2);
            async_cp16_o16(la, gA + (size_t)nk * 2);
            async_cp16(lb, gB + (size_t)nk * 2);
            wait_async3();
        } else {
            wait_async0();
        }
        __syncthreads();

        const unsigned* Al = &As[buf][0];
        const unsigned* Bl = &Bs[buf][0];
        BF16Frag af[2], bfr[2];
#pragma unroll
        for (int mt = 0; mt < 2; ++mt) {
            int row = wm * 32 + mt * 16 + lr;
#pragma unroll
            for (int v = 0; v < 8; ++v) {
                int k = ((v >> 2) << 4) + ((v & 3) << 1) + (g << 3);
                af[mt].u[v] = Al[row * 20 + (k >> 1)];
            }
        }
#pragma unroll
        for (int nt = 0; nt < 2; ++nt) {
            int col = wn * 32 + nt * 16 + lr;
#pragma unroll
            for (int v = 0; v < 8; ++v) {
                int k = ((v >> 2) << 4) + ((v & 3) << 1) + (g << 3);
                bfr[nt].u[v] = Bl[col * 20 + (k >> 1)];
            }
        }
#pragma unroll
        for (int mt = 0; mt < 2; ++mt)
#pragma unroll
            for (int nt = 0; nt < 2; ++nt)
                acc[mt][nt] = __builtin_amdgcn_wmma_f32_16x16x32_bf16(
                    false, af[mt].v, false, bfr[nt].v, (short)0, acc[mt][nt], false, false);
        __syncthreads();
        buf ^= 1;
    }

#pragma unroll
    for (int mt = 0; mt < 2; ++mt)
#pragma unroll
        for (int nt = 0; nt < 2; ++nt)
#pragma unroll
            for (int c = 0; c < 8; ++c) {
                int row = rowbase + wm * 32 + mt * 16 + c + g * 8;
                int col = colbase + wn * 32 + nt * 16 + lr;
                float v = acc[mt][nt][c];
                if (scale) v *= scale[row];
                if (relu) v = fmaxf(v, 0.0f);
                out[(size_t)row * ldc + coff + col] = v;
            }
}

__global__ void k_count(const int* __restrict__ batch, float* __restrict__ cnt) {
    int i = blockIdx.x * blockDim.x + threadIdx.x;
    if (i < NN) {
        int b = batch[i];
        if ((unsigned)b < NG) atomicAdd(&cnt[b], 1.0f);
    }
}

__global__ void k_pool(const float* __restrict__ h, const int* __restrict__ batch,
                       float* __restrict__ pool) {
    long long idx = blockIdx.x * (long long)blockDim.x + threadIdx.x;
    long long total = (long long)NN * OUTD;
    if (idx >= total) return;
    int i = (int)(idx / OUTD), j = (int)(idx % OUTD);
    int b = batch[i];
    if ((unsigned)b < NG) atomicAdd(&pool[(size_t)b * OUTD + j], h[idx]);
}

__global__ __launch_bounds__(256)
void k_final(const float* __restrict__ pool, const float* __restrict__ cnt,
             const float* __restrict__ Wo, const float* __restrict__ bo,
             float* __restrict__ out, int out_size) {
    __shared__ float mean[OUTD];
    int gidx = blockIdx.x, tid = threadIdx.x;
    float ic = 1.0f / fmaxf(cnt[gidx], 1.0f);
    for (int k = tid; k < OUTD; k += 256) mean[k] = pool[(size_t)gidx * OUTD + k] * ic;
    __syncthreads();
    float acc = bo[tid];
    for (int k = 0; k < OUTD; ++k) acc += mean[k] * Wo[(size_t)k * HID + tid];
    out[gidx * HID + tid] = acc;
    if (gidx == 0 && tid == 0 && out_size > NG * HID) out[NG * HID] = 0.0f;
}

extern "C" void kernel_launch(void* const* d_in, const int* in_sizes, int n_in,
                              void* d_out, int out_size, void* d_ws, size_t ws_size,
                              hipStream_t stream) {
    (void)in_sizes; (void)n_in; (void)ws_size;
    const float* x       = (const float*)d_in[0];
    const int*   edges   = (const int*)d_in[1];
    const int*   batch   = (const int*)d_in[2];
    const float* w_embed = (const float*)d_in[3];
    const float* W_out   = (const float*)d_in[4];
    const float* b_out   = (const float*)d_in[5];
    float* out = (float*)d_out;

    char* ws = (char*)d_ws;
    size_t off = 0;
    auto alloc = [&](size_t bytes) -> char* {
        char* p = ws + off;
        off += (bytes + 255) & ~(size_t)255;
        return p;
    };
    unsigned char*  Ab      = (unsigned char*)alloc((size_t)NN * NN);
    unsigned char*  AbT     = (unsigned char*)alloc((size_t)NN * NN);
    unsigned*       selfdup = (unsigned*)alloc((size_t)NN * 4);
    unsigned short* A1      = (unsigned short*)alloc((size_t)NN * NN * 2);
    unsigned short* A2      = (unsigned short*)alloc((size_t)NN * NN * 2);
    float*          dinv1   = (float*)alloc((size_t)NN * 4);
    float*          dinv2   = (float*)alloc((size_t)NN * 4);
    float*          h       = (float*)alloc((size_t)NN * OUTD * 4);
    unsigned short* rsT1    = (unsigned short*)alloc((size_t)512 * NN * 2);
    unsigned short* rsT2    = (unsigned short*)alloc((size_t)512 * NN * 2);
    unsigned short* xb      = (unsigned short*)alloc((size_t)NN * FEAT * 2);
    unsigned short* wbT     = (unsigned short*)alloc((size_t)HID * FEAT * 2);
    float*          pool    = (float*)alloc((size_t)NG * OUTD * 4);
    float*          cnt     = (float*)alloc((size_t)NG * 4);

    k_zero_u32<<<4096, 256, 0, stream>>>((unsigned*)Ab,  (long long)NN * NN / 4);
    k_zero_u32<<<4096, 256, 0, stream>>>((unsigned*)AbT, (long long)NN * NN / 4);
    k_zero_u32<<<32, 256, 0, stream>>>(selfdup, NN);
    k_zero_u32<<<512, 256, 0, stream>>>((unsigned*)pool, (long long)NG * OUTD);
    k_zero_u32<<<1, 64, 0, stream>>>((unsigned*)cnt, NG);
    k_scatter<<<NE / 256, 256, 0, stream>>>(edges, Ab, AbT, selfdup);

    k_adj2<<<dim3(NN / 128, NN / 64), 256, 0, stream>>>(Ab, AbT, selfdup, A1, A2);
    k_degree<<<NN, 256, 0, stream>>>(A1, A2, dinv1, dinv2);

    k_cast_bf16<<<(int)(((long long)NN * FEAT + 255) / 256), 256, 0, stream>>>(x, xb, (long long)NN * FEAT);
    k_castT_w<<<(HID * FEAT + 255) / 256, 256, 0, stream>>>(w_embed, wbT);
    k_gemm_bf16<<<dim3(NN / 128, HID / 64), 256, 0, stream>>>(xb, wbT, h, FEAT, HID, OUTD, 0, nullptr, 1);

    long long n256 = (long long)NN * HID;
    k_scale_castT<<<(int)((n256 + 255) / 256), 256, 0, stream>>>(h, dinv1, rsT1, HID, 0);
    k_scale_castT<<<(int)((n256 + 255) / 256), 256, 0, stream>>>(h, dinv2, rsT2, HID, 0);
    k_gemm_bf16<<<dim3(NN / 128, HID / 64), 256, 0, stream>>>(A1, rsT1, h, NN, HID, OUTD, 256, dinv1, 1);
    k_gemm_bf16<<<dim3(NN / 128, HID / 64), 256, 0, stream>>>(A2, rsT2, h, NN, HID, OUTD, 512, dinv2, 1);

    long long n512 = (long long)NN * 512;
    k_scale_castT<<<(int)((n512 + 255) / 256), 256, 0, stream>>>(h, dinv1, rsT1, 512, 256);
    k_scale_castT<<<(int)((n512 + 255) / 256), 256, 0, stream>>>(h, dinv2, rsT2, 512, 256);
    k_gemm_bf16<<<dim3(NN / 128, 512 / 64), 256, 0, stream>>>(A1, rsT1, h, NN, 512, OUTD, 768, dinv1, 1);
    k_gemm_bf16<<<dim3(NN / 128, 512 / 64), 256, 0, stream>>>(A2, rsT2, h, NN, 512, OUTD, 1280, dinv2, 1);

    k_count<<<NN / 256, 256, 0, stream>>>(batch, cnt);
    k_pool<<<(int)(((long long)NN * OUTD + 255) / 256), 256, 0, stream>>>(h, batch, pool);
    k_final<<<NG, 256, 0, stream>>>(pool, cnt, W_out, b_out, out, out_size);
}
